// MultiHeadedAttention_56324201119967
// MI455X (gfx1250) — matrix operand order repaired and checked by reading, not compiled, not run on hardware
//
#include <hip/hip_runtime.h>

// ---------------------------------------------------------------------------
// Types for CDNA5 WMMA (wave32): v_wmma_f32_16x16x32_bf16
// ---------------------------------------------------------------------------
typedef unsigned short u16;
typedef __attribute__((ext_vector_type(8)))  unsigned short u16x8;
typedef __attribute__((ext_vector_type(16))) __bf16         v16bf;
typedef __attribute__((ext_vector_type(8)))  float          v8f;

// Native f32 -> bf16 conversion (clang lowers to v_cvt_*bf16_f32 on gfx1250,
// vectorizing pairs into the packed form where possible).
static __device__ inline u16 f2bf(float f) {
  return __builtin_bit_cast(u16, (__bf16)f);
}
static __device__ inline unsigned pk2bf(float lo, float hi) {
  return (unsigned)f2bf(lo) | ((unsigned)f2bf(hi) << 16);
}

union FragU {
  struct { u16x8 lo, hi; } h;
  v16bf v;
};
static __device__ inline v16bf mkfrag(u16x8 lo, u16x8 hi) {
  FragU u; u.h.lo = lo; u.h.hi = hi; return u.v;
}
static __device__ inline v8f zero8() {
  v8f z;
#pragma unroll
  for (int i = 0; i < 8; ++i) z[i] = 0.f;
  return z;
}

#define DMODEL 1024

// ---------------------------------------------------------------------------
// GEMM: Y[m,n] = sum_k W[m,k] * X[k,n] + bias[m] (+ R[m,n])
// M = N = K = 1024.  Block: 256 threads (8 waves). Tile: 128x128, BK=32.
// Wave w owns rows [w*16, w*16+16): 8 WMMA tiles of 16x16 along n.
// LDS layouts are pre-swizzled so each bf16 fragment is two 16B ds loads;
// staging uses packed b128/b32 stores.
// ---------------------------------------------------------------------------
__global__ __launch_bounds__(256) void gemm_bias_res(
    const float* __restrict__ W, const float* __restrict__ X,
    const float* __restrict__ bias, const float* __restrict__ R,
    float* __restrict__ Y) {
  __shared__ alignas(16) u16 sA[128 * 40];  // [row m][k], stride 40
  __shared__ alignas(16) u16 sB[128 * 40];  // [col n][k], stride 40

  const int tid  = threadIdx.x;
  const int wave = tid >> 5;
  const int lane = tid & 31;
  const int half = lane >> 4;
  const int lr   = lane & 15;
  const int m0   = blockIdx.y * 128;
  const int n0   = blockIdx.x * 128;

  v8f acc[8];
#pragma unroll
  for (int t = 0; t < 8; ++t) acc[t] = zero8();

  for (int k0 = 0; k0 < DMODEL; k0 += 32) {
    // ---- stage A tile (128x32 of W), row-major [m][k]: 2 x ds_store_b128 ----
    {
      const int r  = tid >> 1;
      const int kb = (tid & 1) * 16;
      const float* src = W + (size_t)(m0 + r) * DMODEL + k0 + kb;
      if (k0 + 32 < DMODEL)
        __builtin_prefetch(src + 32, 0, 1);   // global_prefetch for next K-tile
      const float4 f0 = *(const float4*)(src + 0);
      const float4 f1 = *(const float4*)(src + 4);
      const float4 f2 = *(const float4*)(src + 8);
      const float4 f3 = *(const float4*)(src + 12);
      u16x8 p0, p1;
      p0[0] = f2bf(f0.x); p0[1] = f2bf(f0.y); p0[2] = f2bf(f0.z); p0[3] = f2bf(f0.w);
      p0[4] = f2bf(f1.x); p0[5] = f2bf(f1.y); p0[6] = f2bf(f1.z); p0[7] = f2bf(f1.w);
      p1[0] = f2bf(f2.x); p1[1] = f2bf(f2.y); p1[2] = f2bf(f2.z); p1[3] = f2bf(f2.w);
      p1[4] = f2bf(f3.x); p1[5] = f2bf(f3.y); p1[6] = f2bf(f3.z); p1[7] = f2bf(f3.w);
      *(u16x8*)&sA[r * 40 + kb]     = p0;
      *(u16x8*)&sA[r * 40 + kb + 8] = p1;
    }
    // ---- stage B tile (32x128 of X), transposed to [n][k]: packed b32 ----
#pragma unroll
    for (int i = 0; i < 8; ++i) {
      const int idx = tid + 256 * i;          // 0..2047
      const int kp  = idx >> 7;               // k-pair 0..15
      const int nn  = idx & 127;
      const float x0 = X[(size_t)(k0 + 2 * kp)     * DMODEL + n0 + nn];
      const float x1 = X[(size_t)(k0 + 2 * kp + 1) * DMODEL + n0 + nn];
      *(unsigned*)&sB[nn * 40 + 2 * kp] = pk2bf(x0, x1);
    }
    __syncthreads();

    // A fragment: lane lr row, K halves at half*8 and 16+half*8
    const u16* ap = &sA[(wave * 16 + lr) * 40 + half * 8];
    const v16bf a = mkfrag(*(const u16x8*)ap, *(const u16x8*)(ap + 16));
    // preload all 8 B fragments, then issue the 8 WMMAs back-to-back
    v16bf b[8];
#pragma unroll
    for (int t = 0; t < 8; ++t) {
      const u16* bp = &sB[(t * 16 + lr) * 40 + half * 8];
      b[t] = mkfrag(*(const u16x8*)bp, *(const u16x8*)(bp + 16));
    }
#pragma unroll
    for (int t = 0; t < 8; ++t)
      acc[t] = __builtin_amdgcn_wmma_f32_16x16x32_bf16(
          false, a, false, b[t], (short)0, acc[t], false, false);
    __syncthreads();
  }

  // ---- epilogue: bias (+ residual) ----
  float bl[8];
#pragma unroll
  for (int r = 0; r < 8; ++r) bl[r] = bias[m0 + wave * 16 + half * 8 + r];
#pragma unroll
  for (int t = 0; t < 8; ++t) {
#pragma unroll
    for (int r = 0; r < 8; ++r) {
      const int m = m0 + wave * 16 + half * 8 + r;
      const int n = n0 + t * 16 + lr;
      float v = acc[t][r] + bl[r];
      if (R) v += R[(size_t)m * DMODEL + n];
      Y[(size_t)m * DMODEL + n] = v;
    }
  }
}

// ---------------------------------------------------------------------------
// Attention core, one wave per (d, 16-wide n-tile).
//   S[n,m] = (1/8) * sum_h Q[d,h,n] K[d,h,m]   (h contraction padded 16->32)
//   P = softmax_m(S)  (flash/online over 128-wide m-chunks)
//   X[d,h,n] = sum_m V[d,h,m] P[n,m]
// WMMA C layout gives each lane a fixed output column -> flash rescale is a
// uniform scalar multiply of the 8 accumulator VGPRs per lane.
// ---------------------------------------------------------------------------
__global__ __launch_bounds__(32) void attn_core(
    const float* __restrict__ Qf, const float* __restrict__ Kf,
    const float* __restrict__ Vf, float* __restrict__ Xf) {
  __shared__ alignas(16) u16   sQ[16 * 40];    // A for S-gemm: [n][h], k-pad 16..31 = 0
  __shared__ alignas(16) u16   sK[128 * 40];   // B for S-gemm: [m][h], k-pad = 0
  __shared__ alignas(16) float sS[16 * 132];   // scores chunk [n][m_local]
  __shared__ alignas(16) u16   sP[16 * 136];   // B for X-gemm: [n][m_local]
  __shared__ alignas(16) u16   sV[16 * 136];   // A for X-gemm: [h][m_local]

  const int lane = threadIdx.x;
  const int half = lane >> 4;
  const int lr   = lane & 15;
  const int d    = blockIdx.y;
  const int n0   = blockIdx.x * 16;

  // zero K-dim padding (written once; chunks only touch k<16) + load Q
  for (int idx = lane; idx < 256; idx += 32) {
    const int h = idx >> 4, n = idx & 15;
    sQ[n * 40 + 16 + h] = 0;
    sQ[n * 40 + h] = f2bf(Qf[(size_t)(d * 16 + h) * DMODEL + n0 + n]);
  }
  for (int idx = lane; idx < 1024; idx += 32) {
    const int kp = idx >> 7, mj = idx & 127;   // pad pairs 16..31
    *(unsigned*)&sK[mj * 40 + 16 + 2 * kp] = 0u;
  }
  __syncthreads();

  const u16* qp = &sQ[lr * 40 + half * 8];
  const v16bf aQ = mkfrag(*(const u16x8*)qp, *(const u16x8*)(qp + 16));

  float rowM = -3.0e38f, rowL = 0.f;
  v8f accX = zero8();

  for (int c = 0; c < 8; ++c) {
    const int mb = c * 128;
    // stage K chunk as [m][h] (pack 2 h per b32) and V chunk as [h][m]
    // (pack 2 m per b32, float2 global loads)
    for (int idx = lane; idx < 1024; idx += 32) {
      const int hp = idx >> 7;                 // h-pair 0..7
      const int mj = idx & 127;
      const size_t gk = (size_t)(d * 16 + 2 * hp) * DMODEL + mb + mj;
      *(unsigned*)&sK[mj * 40 + 2 * hp] = pk2bf(Kf[gk], Kf[gk + DMODEL]);
      const int h  = idx >> 6;                 // 0..15
      const int mp = idx & 63;                 // m-pair 0..63
      const float2 v2 =
          *(const float2*)&Vf[(size_t)(d * 16 + h) * DMODEL + mb + 2 * mp];
      *(unsigned*)&sV[h * 136 + 2 * mp] = pk2bf(v2.x, v2.y);
    }
    __syncthreads();

    // scores: preload 8 B fragments, then 8 WMMAs; results -> sS (scaled)
    v16bf bk[8];
#pragma unroll
    for (int t = 0; t < 8; ++t) {
      const u16* bp = &sK[(t * 16 + lr) * 40 + half * 8];
      bk[t] = mkfrag(*(const u16x8*)bp, *(const u16x8*)(bp + 16));
    }
#pragma unroll
    for (int t = 0; t < 8; ++t) {
      v8f s = __builtin_amdgcn_wmma_f32_16x16x32_bf16(
          false, aQ, false, bk[t], (short)0, zero8(), false, false);
#pragma unroll
      for (int r = 0; r < 8; ++r)
        sS[(half * 8 + r) * 132 + t * 16 + lr] = s[r] * 0.125f;
    }
    __syncthreads();

    // online softmax stats for row lr (redundant on both halves), b128 reads
    const float4* srow = (const float4*)&sS[lr * 132];
    float cmax = -3.0e38f;
#pragma unroll 4
    for (int j = 0; j < 32; ++j) {
      const float4 s4 = srow[j];
      cmax = fmaxf(cmax, fmaxf(fmaxf(s4.x, s4.y), fmaxf(s4.z, s4.w)));
    }
    const float newM = fmaxf(rowM, cmax);
    const float corr = __expf(rowM - newM);
    float csum = 0.f;
#pragma unroll 4
    for (int j = 0; j < 32; ++j) {
      const float4 s4 = srow[j];
      const float p0 = __expf(s4.x - newM), p1 = __expf(s4.y - newM);
      const float p2 = __expf(s4.z - newM), p3 = __expf(s4.w - newM);
      csum += (p0 + p1) + (p2 + p3);
      if (half == 0) {
        uint2 pk;
        pk.x = pk2bf(p0, p1);
        pk.y = pk2bf(p2, p3);
        *(uint2*)&sP[lr * 136 + 4 * j] = pk;   // 8B-aligned packed store
      }
    }
    rowL = rowL * corr + csum;
    rowM = newM;
#pragma unroll
    for (int r = 0; r < 8; ++r) accX[r] *= corr;
    __syncthreads();

    // X += V_chunk (16h x 128m) * P_chunk^T (128m x 16n): 4 WMMAs (K=32)
    v16bf av[4], bp4[4];
#pragma unroll
    for (int u = 0; u < 4; ++u) {
      const u16* va = &sV[lr * 136 + u * 32 + half * 8];
      av[u] = mkfrag(*(const u16x8*)va, *(const u16x8*)(va + 16));
      const u16* pb = &sP[lr * 136 + u * 32 + half * 8];
      bp4[u] = mkfrag(*(const u16x8*)pb, *(const u16x8*)(pb + 16));
    }
#pragma unroll
    for (int u = 0; u < 4; ++u)
      accX = __builtin_amdgcn_wmma_f32_16x16x32_bf16(
          false, av[u], false, bp4[u], (short)0, accX, false, false);
    __syncthreads();
  }

  const float inv = 1.0f / rowL;
#pragma unroll
  for (int r = 0; r < 8; ++r)
    Xf[(size_t)(d * 16 + half * 8 + r) * DMODEL + n0 + lr] = accX[r] * inv;
}

// ---------------------------------------------------------------------------
// Host orchestration: 4 sequential attention calls, each = 3 projections,
// attention core, output projection fused with bias+residual.
// ---------------------------------------------------------------------------
extern "C" void kernel_launch(void* const* d_in, const int* in_sizes, int n_in,
                              void* d_out, int out_size, void* d_ws, size_t ws_size,
                              hipStream_t stream) {
  (void)in_sizes; (void)n_in; (void)out_size; (void)ws_size;
  const float* desc0 = (const float*)d_in[0];
  const float* desc1 = (const float*)d_in[1];
  const float* Wq = (const float*)d_in[2]; const float* bq = (const float*)d_in[3];
  const float* Wk = (const float*)d_in[4]; const float* bk = (const float*)d_in[5];
  const float* Wv = (const float*)d_in[6]; const float* bv = (const float*)d_in[7];
  const float* Wm = (const float*)d_in[8]; const float* bm = (const float*)d_in[9];

  const size_t MAT = (size_t)DMODEL * DMODEL;
  float* ws   = (float*)d_ws;
  float* Qb   = ws + 0 * MAT;
  float* Kb   = ws + 1 * MAT;
  float* Vb   = ws + 2 * MAT;
  float* Xb   = ws + 3 * MAT;
  float* cur0 = ws + 4 * MAT;
  float* cur1 = ws + 5 * MAT;
  float* out0 = (float*)d_out;
  float* out1 = out0 + MAT;

  const dim3 gg(8, 8), gb(256);
  const dim3 ag(64, 64), ab(32);

  auto attn = [&](const float* qin, const float* kvin, const float* res, float* out) {
    gemm_bias_res<<<gg, gb, 0, stream>>>(Wq, qin,  bq, nullptr, Qb);
    gemm_bias_res<<<gg, gb, 0, stream>>>(Wk, kvin, bk, nullptr, Kb);
    gemm_bias_res<<<gg, gb, 0, stream>>>(Wv, kvin, bv, nullptr, Vb);
    attn_core   <<<ag, ab, 0, stream>>>(Qb, Kb, Vb, Xb);
    gemm_bias_res<<<gg, gb, 0, stream>>>(Wm, Xb, bm, res, out);
  };

  attn(desc0, desc0, desc0, cur0);  // d0 = desc0 + attn(desc0,desc0,desc0)
  attn(desc1, desc1, desc1, cur1);  // d1 = desc1 + attn(desc1,desc1,desc1)
  attn(cur0,  cur1,  cur0,  out0);  // d0 = d0 + attn(d0, d1, d1)
  attn(cur1,  out0,  cur1,  out1);  // d1 = d1 + attn(d1, d0, d0)
}
